// AttentionWithALiBi_83451214561952
// MI455X (gfx1250) — matrix operand order repaired and checked by reading, not compiled, not run on hardware
//
#include <hip/hip_runtime.h>
#include <math.h>

typedef __attribute__((ext_vector_type(16))) _Float16 v16h;
typedef __attribute__((ext_vector_type(4)))  _Float16 v4h;
typedef __attribute__((ext_vector_type(8)))  float    v8f;

#define WMMA_F32_F16(a, b, c) \
    __builtin_amdgcn_wmma_f32_16x16x32_f16(false, (a), false, (b), (short)0, (c), false, false)

namespace {
constexpr int HEADS = 16;
constexpr int SEQ   = 2048;
constexpr int DIM   = 64;
constexpr int MTILE = 64;   // query rows per block (4 waves x 16)
constexpr int NTILE = 32;   // kv rows per iteration
constexpr int KST   = 72;   // LDS row stride (halves) for K tile, padded
constexpr int VST   = 40;   // LDS row stride (halves) for V^T tile, padded
constexpr int PST   = 34;   // LDS row stride (halves) for P slab, padded
}

// fetch one 32x64 fp32 tile of K and V into registers (4 float4 each)
__device__ __forceinline__ void load_tile(const float* __restrict__ Kp,
                                          const float* __restrict__ Vp,
                                          int tid, float4 kr[4], float4 vr[4])
{
#pragma unroll
    for (int i = 0; i < 4; ++i) {
        int idx = tid + 128 * i;                 // quad index, 512 quads total
        int r = idx >> 4, c4 = (idx & 15) << 2;
        kr[i] = *(const float4*)(Kp + r * DIM + c4);
        vr[i] = *(const float4*)(Vp + r * DIM + c4);
    }
}

// convert + store registers into an LDS buffer (K row-major f16, V transposed f16)
__device__ __forceinline__ void store_tile(_Float16* __restrict__ Kbuf,
                                           _Float16* __restrict__ Vbuf,
                                           int tid, const float4 kr[4], const float4 vr[4])
{
#pragma unroll
    for (int i = 0; i < 4; ++i) {
        int idx = tid + 128 * i;
        int r = idx >> 4, c4 = (idx & 15) << 2;
        v4h kh;
        kh[0] = (_Float16)kr[i].x; kh[1] = (_Float16)kr[i].y;
        kh[2] = (_Float16)kr[i].z; kh[3] = (_Float16)kr[i].w;
        *(v4h*)(Kbuf + r * KST + c4) = kh;       // 8B aligned: KST even, c4 % 4 == 0
        Vbuf[(c4 + 0) * VST + r] = (_Float16)vr[i].x;
        Vbuf[(c4 + 1) * VST + r] = (_Float16)vr[i].y;
        Vbuf[(c4 + 2) * VST + r] = (_Float16)vr[i].z;
        Vbuf[(c4 + 3) * VST + r] = (_Float16)vr[i].w;
    }
}

__global__ __launch_bounds__(128)
void fa_alibi_wmma(const float* __restrict__ Q, const float* __restrict__ K,
                   const float* __restrict__ V, float* __restrict__ O)
{
    __shared__ _Float16 Ksh[2][NTILE * KST];   // double-buffered K tile [kv][d]
    __shared__ _Float16 Vtsh[2][DIM * VST];    // double-buffered V^T tile [d][kv]
    __shared__ _Float16 Psh[4][16 * PST];      // per-wave P staging [row][kv]

    const int tid  = threadIdx.x;
    const int lane = tid & 31;
    const int wave = tid >> 5;
    const int m    = lane & 15;   // col in B/C frags, row in A frags
    const int hi   = lane >> 4;   // 16-lane half

    const int head = blockIdx.x >> 5;            // 32 q-tiles per head
    const int q0   = (blockIdx.x & 31) * MTILE;
    const int qrow = q0 + wave * 16;

    const float scale = 0.125f;                              // 1/sqrt(64)
    const float slope = exp2f(-0.5f * (float)(head + 1));    // (2^(-8/H))^(h+1)

    const float* Kbase = K + (size_t)head * SEQ * DIM;
    const float* Vbase = V + (size_t)head * SEQ * DIM;

    // ---- load Q A-fragments (16x64 -> two K=32 groups, f16) ----
    v16h qa0, qa1;
    {
        const float* qp = Q + ((size_t)(head * SEQ + qrow + m)) * DIM;
#pragma unroll
        for (int j = 0; j < 8; ++j) {
            int k0 = 2 * j + (hi ? 16 : 0);   // A-layout map
            qa0[2 * j]     = (_Float16)qp[k0];
            qa0[2 * j + 1] = (_Float16)qp[k0 + 1];
            qa1[2 * j]     = (_Float16)qp[32 + k0];
            qa1[2 * j + 1] = (_Float16)qp[32 + k0 + 1];
        }
    }

    // ---- loop-invariant ALiBi bias (affine in kv0) ----
    float bias0[8], bias1[8];
#pragma unroll
    for (int r = 0; r < 8; ++r) {
        int srow = qrow + r + 8 * hi;
        bias0[r] = slope * (float)(m - srow);
        bias1[r] = slope * (float)(16 + m - srow);
    }

    v8f o0 = {}, o1 = {}, o2 = {}, o3 = {};
    float mrun[8], lrun[8];
#pragma unroll
    for (int r = 0; r < 8; ++r) { mrun[r] = -1e30f; lrun[r] = 0.0f; }

    // ---- prologue: fill buffer 0 ----
    float4 kreg[4], vreg[4];
    load_tile(Kbase, Vbase, tid, kreg, vreg);
    store_tile(Ksh[0], Vtsh[0], tid, kreg, vreg);
    __syncthreads();

    int buf = 0;
    for (int kv0 = 0; kv0 < SEQ; kv0 += NTILE) {
        const bool has_next = (kv0 + NTILE) < SEQ;
        // issue next tile's global loads before touching this tile's LDS
        if (has_next)
            load_tile(Kbase + (size_t)(kv0 + NTILE) * DIM,
                      Vbase + (size_t)(kv0 + NTILE) * DIM, tid, kreg, vreg);
        if (kv0 + 2 * NTILE < SEQ) {
            size_t pg = (size_t)(kv0 + 2 * NTILE) * DIM + (size_t)tid * 16;
            __builtin_prefetch(Kbase + pg, 0, 0);
            __builtin_prefetch(Vbase + pg, 0, 0);
        }

        const _Float16* __restrict__ Kc = Ksh[buf];
        const _Float16* __restrict__ Vc = Vtsh[buf];

        // ---- S = Q K^T, two 16-col tiles ----
        v8f s0 = {}, s1 = {};
        {
            v16h b;
#pragma unroll
            for (int j = 0; j < 8; ++j) { int k0 = 2 * j + (hi ? 16 : 0);
                b[2 * j] = Kc[m * KST + k0]; b[2 * j + 1] = Kc[m * KST + k0 + 1]; }
            s0 = WMMA_F32_F16(qa0, b, s0);
#pragma unroll
            for (int j = 0; j < 8; ++j) { int k0 = 32 + 2 * j + (hi ? 16 : 0);
                b[2 * j] = Kc[m * KST + k0]; b[2 * j + 1] = Kc[m * KST + k0 + 1]; }
            s0 = WMMA_F32_F16(qa1, b, s0);
#pragma unroll
            for (int j = 0; j < 8; ++j) { int k0 = 2 * j + (hi ? 16 : 0);
                b[2 * j] = Kc[(16 + m) * KST + k0]; b[2 * j + 1] = Kc[(16 + m) * KST + k0 + 1]; }
            s1 = WMMA_F32_F16(qa0, b, s1);
#pragma unroll
            for (int j = 0; j < 8; ++j) { int k0 = 32 + 2 * j + (hi ? 16 : 0);
                b[2 * j] = Kc[(16 + m) * KST + k0]; b[2 * j + 1] = Kc[(16 + m) * KST + k0 + 1]; }
            s1 = WMMA_F32_F16(qa1, b, s1);
        }

        // ---- scale + ALiBi (hoisted bias + per-iter scalar), row-max ----
        const float sb = slope * (float)kv0;
        float rmax[8];
#pragma unroll
        for (int r = 0; r < 8; ++r) {
            s0[r] = fmaf(s0[r], scale, bias0[r] + sb);
            s1[r] = fmaf(s1[r], scale, bias1[r] + sb);
            float rm = fmaxf(s0[r], s1[r]);
            rm = fmaxf(rm, __shfl_xor(rm, 1));
            rm = fmaxf(rm, __shfl_xor(rm, 2));
            rm = fmaxf(rm, __shfl_xor(rm, 4));
            rm = fmaxf(rm, __shfl_xor(rm, 8));
            rmax[r] = rm;
        }

        // ---- online softmax update + stage P ----
#pragma unroll
        for (int r = 0; r < 8; ++r) {
            float mnew  = fmaxf(mrun[r], rmax[r]);
            float alpha = __expf(mrun[r] - mnew);
            mrun[r] = mnew;
            float p0 = __expf(s0[r] - mnew);
            float p1 = __expf(s1[r] - mnew);
            float rs = p0 + p1;
            rs += __shfl_xor(rs, 1);
            rs += __shfl_xor(rs, 2);
            rs += __shfl_xor(rs, 4);
            rs += __shfl_xor(rs, 8);
            lrun[r] = lrun[r] * alpha + rs;
            o0[r] *= alpha; o1[r] *= alpha; o2[r] *= alpha; o3[r] *= alpha;
            int prow = r + 8 * hi;
            Psh[wave][prow * PST + m]      = (_Float16)p0;
            Psh[wave][prow * PST + 16 + m] = (_Float16)p1;
        }

        // ---- reload P in A-layout (same-wave LDS ops are in order) ----
        v16h pa;
#pragma unroll
        for (int j = 0; j < 8; ++j) {
            int k0 = 2 * j + (hi ? 16 : 0);
            pa[2 * j]     = Psh[wave][m * PST + k0];
            pa[2 * j + 1] = Psh[wave][m * PST + k0 + 1];
        }

        // ---- O += P @ V over four 16-col head-dim tiles ----
        {
            v16h b;
#pragma unroll
            for (int j = 0; j < 8; ++j) { int k0 = 2 * j + (hi ? 16 : 0);
                b[2 * j] = Vc[m * VST + k0]; b[2 * j + 1] = Vc[m * VST + k0 + 1]; }
            o0 = WMMA_F32_F16(pa, b, o0);
#pragma unroll
            for (int j = 0; j < 8; ++j) { int k0 = 2 * j + (hi ? 16 : 0);
                b[2 * j] = Vc[(16 + m) * VST + k0]; b[2 * j + 1] = Vc[(16 + m) * VST + k0 + 1]; }
            o1 = WMMA_F32_F16(pa, b, o1);
#pragma unroll
            for (int j = 0; j < 8; ++j) { int k0 = 2 * j + (hi ? 16 : 0);
                b[2 * j] = Vc[(32 + m) * VST + k0]; b[2 * j + 1] = Vc[(32 + m) * VST + k0 + 1]; }
            o2 = WMMA_F32_F16(pa, b, o2);
#pragma unroll
            for (int j = 0; j < 8; ++j) { int k0 = 2 * j + (hi ? 16 : 0);
                b[2 * j] = Vc[(48 + m) * VST + k0]; b[2 * j + 1] = Vc[(48 + m) * VST + k0 + 1]; }
            o3 = WMMA_F32_F16(pa, b, o3);
        }

        // ---- fill the other buffer for next iteration ----
        if (has_next)
            store_tile(Ksh[buf ^ 1], Vtsh[buf ^ 1], tid, kreg, vreg);
        __syncthreads();
        buf ^= 1;
    }

    // ---- finalize: O / l ----
#pragma unroll
    for (int r = 0; r < 8; ++r) {
        int srow  = qrow + r + 8 * hi;
        float inv = 1.0f / lrun[r];
        size_t base = ((size_t)(head * SEQ + srow)) * DIM + m;
        O[base]      = o0[r] * inv;
        O[base + 16] = o1[r] * inv;
        O[base + 32] = o2[r] * inv;
        O[base + 48] = o3[r] * inv;
    }
}

extern "C" void kernel_launch(void* const* d_in, const int* in_sizes, int n_in,
                              void* d_out, int out_size, void* d_ws, size_t ws_size,
                              hipStream_t stream)
{
    (void)in_sizes; (void)n_in; (void)out_size; (void)d_ws; (void)ws_size;
    const float* q = (const float*)d_in[0];
    const float* k = (const float*)d_in[1];
    const float* v = (const float*)d_in[2];
    float* out = (float*)d_out;

    dim3 grid(HEADS * (SEQ / MTILE));   // 512 workgroups
    dim3 block(128);                    // 4 wave32 per workgroup
    fa_alibi_wmma<<<grid, block, 0, stream>>>(q, k, v, out);
}
